// LSTMDecoder_49340584297158
// MI455X (gfx1250) — matrix operand order repaired and checked by reading, not compiled, not run on hardware
//
#include <hip/hip_runtime.h>

#define B_    4096
#define EMB   256
#define HID   256
#define ID    128
#define G1    1024   // 4*HID
#define G2    512    // 4*ID
#define STEPS 128
#define BT    64     // batch rows per workgroup (4 m-tiles)

typedef __attribute__((ext_vector_type(16))) __bf16        v16bf;
typedef __attribute__((ext_vector_type(8)))  float         v8f;
typedef __attribute__((ext_vector_type(4)))  float         v4f;
typedef __attribute__((ext_vector_type(4)))  unsigned int  u32x4;

union BFrag { v16bf v; u32x4 q[2]; unsigned short s[16]; };
union FFrag { v8f  v; v4f  f[2]; float e[8]; };

__device__ __forceinline__ unsigned short f2bf(float x) {
    unsigned int u = __float_as_uint(x);
    u += 0x7FFFu + ((u >> 16) & 1u);   // round-to-nearest-even
    return (unsigned short)(u >> 16);
}
__device__ __forceinline__ float sigm(float x) { return 1.0f / (1.0f + __expf(-x)); }
__device__ __forceinline__ float tanh_fast(float x) {
    float e = __expf(2.0f * x);
    return (e - 1.0f) / (e + 1.0f);
}

// ---------------------------------------------------------------------------
// Swizzle weight W[R x K] (row-major fp32, gates = h @ W^T) into bf16
// B-fragment layout: frag(nt,kt) -> 32 lanes x 16 bf16; lane l holds
// col n = nt*16 + (l&15), K = kt*32 + (l>>4)*8 + i + (i&8) (table order).
// ---------------------------------------------------------------------------
__global__ void k_wswz(const float* __restrict__ W, unsigned short* __restrict__ Wf, int K) {
    int t    = blockIdx.x * blockDim.x + threadIdx.x;
    int frag = t >> 5;
    int lane = t & 31;
    int KT   = K >> 5;
    int nt   = frag / KT;
    int kt   = frag - nt * KT;
    int n    = nt * 16 + (lane & 15);
    int k    = kt * 32 + (lane >> 4) * 8;
    const float*    p = W  + (size_t)n * K + k;
    unsigned short* d = Wf + ((size_t)frag * 32 + lane) * 16;
#pragma unroll
    for (int i = 0; i < 16; ++i) d[i] = f2bf(p[i + (i & 8)]);
}

// ---------------------------------------------------------------------------
// x1 = emb @ W_ih1^T + b_ih1 + b_hh1, stored as raw D-fragments:
// x1f[(mt*64 + nt)*32 + lane][8 floats].  One wave per 16x16 tile, K = 256.
// ---------------------------------------------------------------------------
__global__ void k_x1(const float* __restrict__ emb, const float* __restrict__ Wih1,
                     const float* __restrict__ bih1, const float* __restrict__ bhh1,
                     float* __restrict__ x1f) {
    int gw   = (blockIdx.x * blockDim.x + threadIdx.x) >> 5; // 0..16383
    int lane = threadIdx.x & 31;
    int mt   = gw >> 6;     // 0..255
    int nt   = gw & 63;     // 0..63
    int lm   = lane & 15, lh = lane >> 4;
    int m    = mt * 16 + lm;
    int n    = nt * 16 + lm;

    v8f acc = {};
#pragma unroll
    for (int kt = 0; kt < 8; ++kt) {
        BFrag a;
        const float* pa = emb + (size_t)m * EMB + kt * 32 + lh * 8;
#pragma unroll
        for (int i = 0; i < 8; ++i) { a.s[i] = f2bf(pa[i]); a.s[8 + i] = f2bf(pa[16 + i]); }
        BFrag b;
        const float* pb = Wih1 + (size_t)n * EMB + kt * 32 + lh * 8;
#pragma unroll
        for (int i = 0; i < 16; ++i) b.s[i] = f2bf(pb[i + (i & 8)]);
        acc = __builtin_amdgcn_wmma_f32_16x16x32_bf16(false, a.v, false, b.v,
                                                      (short)0, acc, false, false);
    }
    float bias = bih1[n] + bhh1[n];
    FFrag o; o.v = acc;
#pragma unroll
    for (int i = 0; i < 8; ++i) o.e[i] += bias;
    float* dst = x1f + ((size_t)gw * 32 + lane) * 8;
    *(v4f*)dst       = o.f[0];
    *(v4f*)(dst + 4) = o.f[1];
}

// ---------------------------------------------------------------------------
// Recurrence: 64 WGs x 64 batch rows, 16 waves (512 thr).  h0/h1 bf16 in LDS,
// c0/c1 in VGPRs.  Gates computed as two accumulator pairs:
// pair0 = (i, g) -> u = sig(i)*tanh(g);  pair1 = (f, o) -> c' = sig(f)*c + u,
// h = sig(o)*tanh(c').  A-fragments are streamed one-at-a-time from LDS and a
// compiler memory barrier between the pairs defeats LDS-load CSE, keeping
// peak VGPR pressure low (no scratch spills).
// ---------------------------------------------------------------------------
__launch_bounds__(512, 1)
__global__ void k_rnn(const float* __restrict__ x1f,
                      const unsigned short* __restrict__ W1f,
                      const unsigned short* __restrict__ W2f,
                      const unsigned short* __restrict__ W3f,
                      const float* __restrict__ bih2, const float* __restrict__ bhh2,
                      float* __restrict__ out) {
    __shared__ __align__(16) unsigned short h0s[BT * HID]; // 32 KB
    __shared__ __align__(16) unsigned short h1s[BT * ID];  // 16 KB

    const int tid   = threadIdx.x;            // 0..511
    const int wave  = tid >> 5, lane = tid & 31;
    const int lm    = lane & 15, lh = lane >> 4;
    const int mhalf = wave >> 3;              // GEMM2: which m-tile pair
    const int nt2   = wave & 7;               // GEMM2: n-tile
    const int wgm   = blockIdx.x * 4;         // global m-tile base (BT/16 = 4)

    for (int i = tid; i < BT * HID; i += 512) h0s[i] = 0;
    for (int i = tid; i < BT * ID;  i += 512) h1s[i] = 0;

    float c0[4][8];   // cell1 state: m-tile, reg
    float c1[2][8];   // cell2 state: local m-tile, reg
#pragma unroll
    for (int a = 0; a < 4; ++a)
#pragma unroll
        for (int r = 0; r < 8; ++r) c0[a][r] = 0.0f;
#pragma unroll
    for (int a = 0; a < 2; ++a)
#pragma unroll
        for (int r = 0; r < 8; ++r) c1[a][r] = 0.0f;

    float b2r[4];     // cell2 bias per gate (lane's column)
#pragma unroll
    for (int g = 0; g < 4; ++g) {
        int n = (nt2 + 8 * g) * 16 + lm;
        b2r[g] = bih2[n] + bhh2[n];
    }
    __syncthreads();

    unsigned int h0pk[4][4]; // new h0, bf16x2 packed
    unsigned int h1pk[2][4]; // new h1, bf16x2 packed

    for (int step = 0; step < STEPS; ++step) {
        // ---------------- cell 1: gates1 = x1 + h0 @ W_hh1^T ----------------
#pragma unroll
        for (int mt = 0; mt < 4; ++mt) {
            float u[8];
#pragma unroll
            for (int pair = 0; pair < 2; ++pair) {
                if (pair == 1) asm volatile("" ::: "memory"); // defeat LDS-load CSE
                // pair 0: gates (i, g) at n-tiles wave, wave+32
                // pair 1: gates (f, o) at n-tiles wave+16, wave+48
                const int n0 = wave + 16 * pair;
                const int n1 = n0 + 32;
                const float* px0 = x1f + ((size_t)((wgm + mt) * 64 + n0) * 32 + lane) * 8;
                const float* px1 = x1f + ((size_t)((wgm + mt) * 64 + n1) * 32 + lane) * 8;
                FFrag c0f; c0f.f[0] = *(const v4f*)px0; c0f.f[1] = *(const v4f*)(px0 + 4);
                FFrag c1f; c1f.f[0] = *(const v4f*)px1; c1f.f[1] = *(const v4f*)(px1 + 4);
                v8f acc0 = c0f.v;
                v8f acc1 = c1f.v;
#pragma unroll
                for (int kt = 0; kt < 8; ++kt) {
                    BFrag a;
                    const unsigned short* p = h0s + (mt * 16 + lm) * HID + kt * 32 + lh * 8;
                    a.q[0] = *(const u32x4*)p;
                    a.q[1] = *(const u32x4*)(p + 16);
                    BFrag b0;
                    const unsigned short* pw0 = W1f + ((size_t)((n0 << 3) + kt) * 32 + lane) * 16;
                    b0.q[0] = *(const u32x4*)pw0;
                    b0.q[1] = *(const u32x4*)(pw0 + 8);
                    acc0 = __builtin_amdgcn_wmma_f32_16x16x32_bf16(false, a.v, false, b0.v,
                                                                   (short)0, acc0, false, false);
                    BFrag b1;
                    const unsigned short* pw1 = W1f + ((size_t)((n1 << 3) + kt) * 32 + lane) * 16;
                    b1.q[0] = *(const u32x4*)pw1;
                    b1.q[1] = *(const u32x4*)(pw1 + 8);
                    acc1 = __builtin_amdgcn_wmma_f32_16x16x32_bf16(false, a.v, false, b1.v,
                                                                   (short)0, acc1, false, false);
                }
                FFrag g0; g0.v = acc0;
                FFrag g1; g1.v = acc1;
                if (pair == 0) {
#pragma unroll
                    for (int r = 0; r < 8; ++r)
                        u[r] = sigm(g0.e[r]) * tanh_fast(g1.e[r]);   // sig(i)*tanh(g)
                } else {
                    float hn[8];
#pragma unroll
                    for (int r = 0; r < 8; ++r) {
                        float c = sigm(g0.e[r]) * c0[mt][r] + u[r];  // sig(f)*c + u
                        c0[mt][r] = c;
                        hn[r] = sigm(g1.e[r]) * tanh_fast(c);        // sig(o)*tanh(c)
                    }
#pragma unroll
                    for (int rp = 0; rp < 4; ++rp)
                        h0pk[mt][rp] = (unsigned int)f2bf(hn[2 * rp]) |
                                       ((unsigned int)f2bf(hn[2 * rp + 1]) << 16);
                }
            }
        }
        __syncthreads();   // all reads of old h0 done
        {
            const int col = wave * 16 + lm;
#pragma unroll
            for (int mt = 0; mt < 4; ++mt) {
#pragma unroll
                for (int rp = 0; rp < 4; ++rp) {
                    unsigned int v = h0pk[mt][rp];
                    int m = mt * 16 + 2 * rp + 8 * lh;
                    h0s[m * HID + col]       = (unsigned short)v;
                    h0s[(m + 1) * HID + col] = (unsigned short)(v >> 16);
                }
            }
        }
        __syncthreads();   // new h0 visible

        // ------- cell 2: gates2 = h0 @ W_ih2^T + h1 @ W_hh2^T + b2 ---------
#pragma unroll
        for (int mi = 0; mi < 2; ++mi) {
            const int mt = mhalf * 2 + mi;
            float u[8];
#pragma unroll
            for (int pair = 0; pair < 2; ++pair) {
                if (pair == 1) asm volatile("" ::: "memory"); // defeat LDS-load CSE
                // pair 0: gates (i, g) at n-tiles nt2, nt2+16
                // pair 1: gates (f, o) at n-tiles nt2+8, nt2+24
                const int n0 = nt2 + 8 * pair;
                const int n1 = n0 + 16;
                const float bb0 = b2r[pair];        // i or f bias
                const float bb1 = b2r[pair + 2];    // g or o bias
                FFrag ci0, ci1;
#pragma unroll
                for (int r = 0; r < 8; ++r) { ci0.e[r] = bb0; ci1.e[r] = bb1; }
                v8f acc0 = ci0.v;
                v8f acc1 = ci1.v;
                // h0 contribution (K = 256)
#pragma unroll
                for (int kt = 0; kt < 8; ++kt) {
                    BFrag a;
                    const unsigned short* p = h0s + (mt * 16 + lm) * HID + kt * 32 + lh * 8;
                    a.q[0] = *(const u32x4*)p;
                    a.q[1] = *(const u32x4*)(p + 16);
                    BFrag b0;
                    const unsigned short* pw0 = W2f + ((size_t)((n0 << 3) + kt) * 32 + lane) * 16;
                    b0.q[0] = *(const u32x4*)pw0;
                    b0.q[1] = *(const u32x4*)(pw0 + 8);
                    acc0 = __builtin_amdgcn_wmma_f32_16x16x32_bf16(false, a.v, false, b0.v,
                                                                   (short)0, acc0, false, false);
                    BFrag b1;
                    const unsigned short* pw1 = W2f + ((size_t)((n1 << 3) + kt) * 32 + lane) * 16;
                    b1.q[0] = *(const u32x4*)pw1;
                    b1.q[1] = *(const u32x4*)(pw1 + 8);
                    acc1 = __builtin_amdgcn_wmma_f32_16x16x32_bf16(false, a.v, false, b1.v,
                                                                   (short)0, acc1, false, false);
                }
                // h1 contribution (K = 128)
#pragma unroll
                for (int kt = 0; kt < 4; ++kt) {
                    BFrag a;
                    const unsigned short* p = h1s + (mt * 16 + lm) * ID + kt * 32 + lh * 8;
                    a.q[0] = *(const u32x4*)p;
                    a.q[1] = *(const u32x4*)(p + 16);
                    BFrag b0;
                    const unsigned short* pw0 = W3f + ((size_t)((n0 << 2) + kt) * 32 + lane) * 16;
                    b0.q[0] = *(const u32x4*)pw0;
                    b0.q[1] = *(const u32x4*)(pw0 + 8);
                    acc0 = __builtin_amdgcn_wmma_f32_16x16x32_bf16(false, a.v, false, b0.v,
                                                                   (short)0, acc0, false, false);
                    BFrag b1;
                    const unsigned short* pw1 = W3f + ((size_t)((n1 << 2) + kt) * 32 + lane) * 16;
                    b1.q[0] = *(const u32x4*)pw1;
                    b1.q[1] = *(const u32x4*)(pw1 + 8);
                    acc1 = __builtin_amdgcn_wmma_f32_16x16x32_bf16(false, a.v, false, b1.v,
                                                                   (short)0, acc1, false, false);
                }
                FFrag g0; g0.v = acc0;
                FFrag g1; g1.v = acc1;
                if (pair == 0) {
#pragma unroll
                    for (int r = 0; r < 8; ++r)
                        u[r] = sigm(g0.e[r]) * tanh_fast(g1.e[r]);
                } else {
                    const int col = nt2 * 16 + lm;
                    float hn[8];
#pragma unroll
                    for (int r = 0; r < 8; ++r) {
                        float c = sigm(g0.e[r]) * c1[mi][r] + u[r];
                        c1[mi][r] = c;
                        float h = sigm(g1.e[r]) * tanh_fast(c);
                        hn[r] = h;
                        int m = mt * 16 + r + 8 * lh;
                        out[(size_t)(blockIdx.x * BT + m) * STEPS * ID + (size_t)step * ID + col] = h;
                    }
#pragma unroll
                    for (int rp = 0; rp < 4; ++rp)
                        h1pk[mi][rp] = (unsigned int)f2bf(hn[2 * rp]) |
                                       ((unsigned int)f2bf(hn[2 * rp + 1]) << 16);
                }
            }
        }
        __syncthreads();   // all reads of old h1 done
        {
            const int col = nt2 * 16 + lm;
#pragma unroll
            for (int mi = 0; mi < 2; ++mi) {
#pragma unroll
                for (int rp = 0; rp < 4; ++rp) {
                    unsigned int v = h1pk[mi][rp];
                    int m = (mhalf * 2 + mi) * 16 + 2 * rp + 8 * lh;
                    h1s[m * ID + col]       = (unsigned short)v;
                    h1s[(m + 1) * ID + col] = (unsigned short)(v >> 16);
                }
            }
        }
        // new-h1 writes are separated from next step's h1 reads by the two
        // barriers at the top of the next iteration -> no extra barrier needed
    }
}

extern "C" void kernel_launch(void* const* d_in, const int* in_sizes, int n_in,
                              void* d_out, int out_size, void* d_ws, size_t ws_size,
                              hipStream_t stream) {
    const float* emb  = (const float*)d_in[0];
    const float* Wih1 = (const float*)d_in[1];
    const float* Whh1 = (const float*)d_in[2];
    const float* bih1 = (const float*)d_in[3];
    const float* bhh1 = (const float*)d_in[4];
    const float* Wih2 = (const float*)d_in[5];
    const float* Whh2 = (const float*)d_in[6];
    const float* bih2 = (const float*)d_in[7];
    const float* bhh2 = (const float*)d_in[8];
    float* out = (float*)d_out;

    char* ws = (char*)d_ws;
    float*          x1f = (float*)ws;                                   // 16 MB
    unsigned short* W1f = (unsigned short*)(ws + (size_t)B_ * G1 * 4);  // 512 KB
    unsigned short* W2f = W1f + (size_t)G1 * HID;                       // 256 KB
    unsigned short* W3f = W2f + (size_t)G2 * HID;                       // 128 KB

    // weight swizzle: (R*K/16) threads each
    k_wswz<<<(G1 * HID / 16) / 256, 256, 0, stream>>>(Whh1, W1f, HID);
    k_wswz<<<(G2 * HID / 16) / 256, 256, 0, stream>>>(Wih2, W2f, HID);
    k_wswz<<<(G2 * ID  / 16) / 256, 256, 0, stream>>>(Whh2, W3f, ID);
    // x1 pre-activation: 16384 waves
    k_x1<<<(16384 * 32) / 256, 256, 0, stream>>>(emb, Wih1, bih1, bhh1, x1f);
    // recurrence: 64 WGs x 16 waves
    k_rnn<<<B_ / BT, 512, 0, stream>>>(x1f, W1f, W2f, W3f, bih2, bhh2, out);
}
